// BertCoAttention_12163347383201
// MI455X (gfx1250) — compile-verified, matrix operand order checked by reading, not run on hardware
//
#include <hip/hip_runtime.h>
#include <hip/hip_bf16.h>

// ---------- types ----------
typedef __attribute__((ext_vector_type(16))) __bf16 v16bf;
typedef __attribute__((ext_vector_type(8)))  __bf16 v8bf;
typedef __attribute__((ext_vector_type(4)))  __bf16 v4bf;
typedef __attribute__((ext_vector_type(8)))  float  v8f;
typedef __attribute__((ext_vector_type(4)))  unsigned int uint32x4;
typedef __attribute__((ext_vector_type(8)))  int  int32x8;
typedef __attribute__((ext_vector_type(4)))  int  int32x4;

#define NUM_HEADS 16
#define HEAD_SIZE 64
#define HIDDEN    1024
#define SEQ       1024
#define BATCH     8
#define HEAD_ELEMS (SEQ * HEAD_SIZE)   // 65536 per (b,h)

// TDM path. The builtin exists on both toolchains but with different arity:
//   ROCm 7.2 / clang-22 : 5 args (g0 v4u32, g1 v8i32, g2 v4i32, g3 v4i32, cpol)
//   amdgpu-toolchain / clang>=23 : 6 args (..., extra v8i32, cpol)
// Dispatch on __clang_major__ instead of disabling one lane.
#if defined(__AMDGCN__) && __has_builtin(__builtin_amdgcn_tensor_load_to_lds)
#define USE_TDM 1
#else
#define USE_TDM 0
#endif

__device__ __forceinline__ v8f zero8() {
  v8f z = {0.f, 0.f, 0.f, 0.f, 0.f, 0.f, 0.f, 0.f};
  return z;
}

__device__ __forceinline__ v16bf combine16(v8bf lo, v8bf hi) {
  return __builtin_shufflevector(lo, hi, 0,1,2,3,4,5,6,7,8,9,10,11,12,13,14,15);
}

// Load one 16x32 (or 32x16) bf16 fragment where the K dimension is contiguous
// in memory. Lane layout (CDNA5 16-bit A/B): lanes 0-15 -> K {kg..kg+7,
// kg+16..kg+23} with kg = 0; lanes 16-31 -> kg = 8.
template <typename P>
__device__ __forceinline__ v16bf load_frag_k16(P rowPtr, int kg) {
  v8bf lo = *(const v8bf*)(rowPtr + kg);
  v8bf hi = *(const v8bf*)(rowPtr + kg + 16);
  return combine16(lo, hi);
}

__device__ __forceinline__ v8f wmma_bf16(v16bf a, v16bf b, v8f c) {
  return __builtin_amdgcn_wmma_f32_16x16x32_bf16(false, a, false, b,
                                                 (short)0, c, false, false);
}

#if USE_TDM
// Issue one TDM 2D tile load: tile = 32 (dim0, contiguous k) x 128 (dim1,
// rows), bf16 (data_size=2B), row stride 1024 elements. Descriptor packing
// per CDNA5 ISA ch.8 (group0: count/lds_addr/global_addr/type=2; group1:
// data_size + dims/strides).
__device__ __forceinline__ void tdm_load_tile(const __bf16* gptr,
                                              const __bf16* lptr,
                                              unsigned int tensorDim1) {
  unsigned long long ga = (unsigned long long)(uintptr_t)gptr;
  unsigned int la = (unsigned int)(uintptr_t)lptr;   // low 32 bits = LDS offset
  uint32x4 g0 = { 1u,                                   // count=1, user mode
                  la,                                   // lds_addr (bytes)
                  (unsigned int)ga,                     // global_addr[31:0]
                  (unsigned int)((ga >> 32) & 0x01FFFFFFu) | (2u << 30) }; // type=2
  int32x8 g1 = { (int)0x00010000u,          // wg_mask=0, data_size=1 (2 bytes)
                 (int)(1024u << 16),        // tensor_dim0 = 1024   (bits 63:48)
                 (int)(tensorDim1 << 16),   // tensor_dim1 [15:0]   (bits 95:80)
                 (int)(32u << 16),          // tile_dim0 = 32       (bits 127:112)
                 (int)128,                  // tile_dim1 = 128      (bits 143:128)
                 (int)1024,                 // tensor_dim0_stride   (bits 191:160)
                 0, 0 };
  int32x4 gz = { 0, 0, 0, 0 };              // 2D tile: groups 2/3 unused
#if __clang_major__ >= 23
  int32x8 gz8 = { 0, 0, 0, 0, 0, 0, 0, 0 };
  __builtin_amdgcn_tensor_load_to_lds(g0, g1, gz, gz, gz8, 0);
#else
  __builtin_amdgcn_tensor_load_to_lds(g0, g1, gz, gz, 0);
#endif
}
#endif

// =====================================================================
// Kernel 0a: fp32 -> bf16 elementwise (8 elems/thread).
// =====================================================================
__global__ __launch_bounds__(256)
void cvt_bf16_kernel(const float* __restrict__ in, __bf16* __restrict__ out) {
  size_t idx = (size_t)(blockIdx.x * 256 + threadIdx.x) * 8;
  float4 a = *(const float4*)(in + idx);
  float4 b = *(const float4*)(in + idx + 4);
  v4bf p0 = { (__bf16)a.x, (__bf16)a.y, (__bf16)a.z, (__bf16)a.w };
  v4bf p1 = { (__bf16)b.x, (__bf16)b.y, (__bf16)b.z, (__bf16)b.w };
  *(v4bf*)(out + idx)     = p0;
  *(v4bf*)(out + idx + 4) = p1;
}

// =====================================================================
// Kernel 0b: W [k][n] fp32 -> Wt [n][k] bf16 (LDS-tiled 32x32 transpose).
// =====================================================================
__global__ __launch_bounds__(256)
void transpose_cvt_kernel(const float* __restrict__ W, __bf16* __restrict__ Wt) {
  __shared__ __bf16 lt[32 * 33];
  const int kBase = blockIdx.y * 32, nBase = blockIdx.x * 32;
  const int t = threadIdx.x;
  const int r = t >> 3, c = (t & 7) * 4;
  float4 v = *(const float4*)(W + (size_t)(kBase + r) * HIDDEN + nBase + c);
  lt[r * 33 + c + 0] = (__bf16)v.x;
  lt[r * 33 + c + 1] = (__bf16)v.y;
  lt[r * 33 + c + 2] = (__bf16)v.z;
  lt[r * 33 + c + 3] = (__bf16)v.w;
  __syncthreads();
  v4bf o = { lt[(c + 0) * 33 + r], lt[(c + 1) * 33 + r],
             lt[(c + 2) * 33 + r], lt[(c + 3) * 33 + r] };
  *(v4bf*)(Wt + (size_t)(nBase + r) * HIDDEN + kBase + c) = o;
}

// =====================================================================
// Kernel 1: out = (Xb @ Wt^T + bias) in bf16, Xb [8192,1024] bf16 (row=k
// contiguous), Wt [n][k] bf16. TDM double-buffered staging.
// Output (m,n) -> out[(b*16+h)*65536 + s*strideS + d*strideD].
//   Q/K: strideS=64, strideD=1  ([b,h,s,d]);  V: strideS=1, strideD=1024.
// Block 256 thr, tile 128x128, BK=32; wave = 32x64 (2x4 WMMA accums).
// =====================================================================
__global__ __launch_bounds__(256)
void qkv_gemm_kernel(const __bf16* __restrict__ Xb, const __bf16* __restrict__ Wt,
                     const float* __restrict__ bias, __bf16* __restrict__ out,
                     int strideS, int strideD)
{
  __shared__ __bf16 lds_a[2][128 * 32];   // [row][k], ping-pong
  __shared__ __bf16 lds_b[2][128 * 32];   // [col][k], ping-pong

  const int tid   = threadIdx.x;
  const int lane  = tid & 31;
  const int w     = tid >> 5;
  const int waveM = w >> 1;            // 0..3 -> 32-row slab
  const int waveN = w & 1;             // 0..1 -> 64-col slab
  const int mBase = blockIdx.y * 128;
  const int nBase = blockIdx.x * 128;
  const int kg    = (lane >> 4) << 3;  // 0 or 8
  const int l15   = lane & 15;

  v8f acc[2][4];
  #pragma unroll
  for (int i = 0; i < 2; ++i)
    #pragma unroll
    for (int j = 0; j < 4; ++j) acc[i][j] = zero8();

#if USE_TDM
  if (w == 0) {   // one wave drives the DMA engine
    tdm_load_tile(Xb + (size_t)mBase * HIDDEN, &lds_a[0][0], 8192u);
    tdm_load_tile(Wt + (size_t)nBase * HIDDEN, &lds_b[0][0], 1024u);
  }
#endif

  for (int kk = 0; kk < HIDDEN; kk += 32) {
    const int cur = (kk >> 5) & 1;
    __syncthreads();   // all readers of buf[cur^1] (prev iter) are done
#if USE_TDM
    if (w == 0) {
      if (kk + 32 < HIDDEN) {
        tdm_load_tile(Xb + (size_t)mBase * HIDDEN + kk + 32, &lds_a[cur ^ 1][0], 8192u);
        tdm_load_tile(Wt + (size_t)nBase * HIDDEN + kk + 32, &lds_b[cur ^ 1][0], 1024u);
        __builtin_amdgcn_s_wait_tensorcnt(2);  // retire current tile pair
      } else {
        __builtin_amdgcn_s_wait_tensorcnt(0);
      }
    }
#else
    // synchronous bf16 staging fallback
    #pragma unroll
    for (int q = 0; q < 2; ++q) {
      int ch = tid + q * 256;              // 512 chunks of 8 bf16 per matrix
      int r = ch >> 2, o8 = (ch & 3) * 8;
      v8bf va = *(const v8bf*)(Xb + (size_t)(mBase + r) * HIDDEN + kk + o8);
      *(v8bf*)(&lds_a[cur][r * 32 + o8]) = va;
      v8bf vb = *(const v8bf*)(Wt + (size_t)(nBase + r) * HIDDEN + kk + o8);
      *(v8bf*)(&lds_b[cur][r * 32 + o8]) = vb;
    }
#endif
    __syncthreads();   // tile `cur` visible to all waves

    v16bf afrag[2], bfrag[4];
    #pragma unroll
    for (int mt = 0; mt < 2; ++mt)
      afrag[mt] = load_frag_k16(&lds_a[cur][(waveM * 32 + mt * 16 + l15) * 32], kg);
    #pragma unroll
    for (int nt = 0; nt < 4; ++nt)
      bfrag[nt] = load_frag_k16(&lds_b[cur][(waveN * 64 + nt * 16 + l15) * 32], kg);

    #pragma unroll
    for (int mt = 0; mt < 2; ++mt)
      #pragma unroll
      for (int nt = 0; nt < 4; ++nt)
        acc[mt][nt] = wmma_bf16(afrag[mt], bfrag[nt], acc[mt][nt]);
  }

  // ---- epilogue: bias add, bf16 store with head-split layout ----
  const int mrow0 = (lane >> 4) << 3;   // D layout: M = j + 8*(lane>>4)
  #pragma unroll
  for (int nt = 0; nt < 4; ++nt) {
    int n = nBase + waveN * 64 + nt * 16 + l15;
    float bv = bias[n];
    int h = n >> 6, d = n & 63;
    #pragma unroll
    for (int mt = 0; mt < 2; ++mt) {
      #pragma unroll
      for (int j = 0; j < 8; ++j) {
        int m = mBase + waveM * 32 + mt * 16 + mrow0 + j;
        int b = m >> 10, s = m & 1023;
        size_t addr = (size_t)(b * NUM_HEADS + h) * HEAD_ELEMS +
                      (size_t)s * strideS + (size_t)d * strideD;
        out[addr] = (__bf16)(acc[mt][nt][j] + bv);
      }
    }
  }
}

// =====================================================================
// Kernel 2: attention for one (b, h, 16-row q tile).
// Q,K: [b,h,s,d] bf16; Vt: [b,h,d,s] bf16; mask: [B,1,1,S2] fp32.
// scores (16x1024 fp32 in LDS) -> softmax -> softmax(1-p+mask) ->
// P@V via WMMA -> fp32 out [B,S1,HIDDEN].
// =====================================================================
__global__ __launch_bounds__(256)
void co_attn_kernel(const __bf16* __restrict__ Q, const __bf16* __restrict__ K,
                    const __bf16* __restrict__ Vt, const float* __restrict__ mask,
                    const int* __restrict__ cl_att, float* __restrict__ out)
{
  __shared__ float sm[16 * SEQ];       // 64 KB score/prob tile
  __shared__ float ctx_acc[16 * 64];   // 4 KB context accumulator

  const int tid  = threadIdx.x;
  const int lane = tid & 31;
  const int w    = tid >> 5;
  const int qT   = blockIdx.x;         // 0..63
  const int h    = blockIdx.y;         // 0..15
  const int b    = blockIdx.z;         // 0..7
  const int bh   = b * NUM_HEADS + h;
  const int kg   = (lane >> 4) << 3;
  const int l15  = lane & 15;
  const size_t headBase = (size_t)bh * HEAD_ELEMS;
  const float* mrow = mask + (size_t)b * SEQ;

  #pragma unroll
  for (int i = 0; i < 4; ++i) ctx_acc[tid + i * 256] = 0.f;

  // Q fragments for this 16-row tile (shared across all score tiles)
  const __bf16* qrow = Q + headBase + (size_t)(qT * 16 + l15) * HEAD_SIZE;
  v16bf a0 = load_frag_k16(qrow, kg);        // d = 0..31
  v16bf a1 = load_frag_k16(qrow + 32, kg);   // d = 32..63

  // ---- scores: wave w covers columns [w*128, w*128+128) ----
  const int mrow0 = (lane >> 4) << 3;
  #pragma unroll
  for (int tt = 0; tt < 8; ++tt) {
    int sCol = w * 128 + tt * 16;
    const __bf16* krow = K + headBase + (size_t)(sCol + l15) * HEAD_SIZE;
    if (tt + 1 < 8) {  // gfx1250 global_prefetch of next K tile row
      __builtin_prefetch(K + headBase + (size_t)(sCol + 16 + l15) * HEAD_SIZE, 0, 0);
    }
    v16bf b0 = load_frag_k16(krow, kg);
    v16bf b1 = load_frag_k16(krow + 32, kg);
    v8f acc = zero8();
    acc = wmma_bf16(a0, b0, acc);
    acc = wmma_bf16(a1, b1, acc);
    float mv = mrow[sCol + l15];
    #pragma unroll
    for (int j = 0; j < 8; ++j)
      sm[(mrow0 + j) * SEQ + sCol + l15] = acc[j] * 0.125f + mv;  // 1/sqrt(64)
  }
  __syncthreads();

  // ---- double softmax, wave w owns rows 2w and 2w+1 ----
  const int do_cl = cl_att[0];
  #pragma unroll
  for (int rr = 0; rr < 2; ++rr) {
    float* row = sm + (w * 2 + rr) * SEQ;
    // softmax #1 over scores(+mask already applied)
    float vmax = -1e30f;
    for (int i = 0; i < 32; ++i) vmax = fmaxf(vmax, row[lane + i * 32]);
    for (int o = 16; o; o >>= 1) vmax = fmaxf(vmax, __shfl_xor(vmax, o, 32));
    float s = 0.f;
    for (int i = 0; i < 32; ++i) {
      int c = lane + i * 32;
      float e = __expf(row[c] - vmax);
      s += e; row[c] = e;
    }
    for (int o = 16; o; o >>= 1) s += __shfl_xor(s, o, 32);
    float inv = 1.f / s;
    for (int i = 0; i < 32; ++i) row[lane + i * 32] *= inv;

    if (do_cl) {
      // softmax #2 over (1 - p + mask)
      float vmax2 = -1e30f;
      for (int i = 0; i < 32; ++i) {
        int c = lane + i * 32;
        vmax2 = fmaxf(vmax2, 1.f - row[c] + mrow[c]);
      }
      for (int o = 16; o; o >>= 1) vmax2 = fmaxf(vmax2, __shfl_xor(vmax2, o, 32));
      float s2 = 0.f;
      for (int i = 0; i < 32; ++i) {
        int c = lane + i * 32;
        float e = __expf(1.f - row[c] + mrow[c] - vmax2);
        s2 += e; row[c] = e;
      }
      for (int o = 16; o; o >>= 1) s2 += __shfl_xor(s2, o, 32);
      float inv2 = 1.f / s2;
      for (int i = 0; i < 32; ++i) row[lane + i * 32] *= inv2;
    }
  }
  __syncthreads();

  // ---- ctx = P @ V : wave w handles K-range [w*128, w*128+128) ----
  v8f cacc[4];
  #pragma unroll
  for (int dt = 0; dt < 4; ++dt) cacc[dt] = zero8();

  #pragma unroll
  for (int ks = 0; ks < 4; ++ks) {
    int kBase = w * 128 + ks * 32;
    // A fragment from fp32 probs in LDS (convert to bf16)
    v16bf ap;
    {
      const float* p = sm + l15 * SEQ + kBase + kg;
      #pragma unroll
      for (int e = 0; e < 8; ++e) ap[e] = (__bf16)p[e];
      #pragma unroll
      for (int e = 0; e < 8; ++e) ap[8 + e] = (__bf16)p[16 + e];
    }
    #pragma unroll
    for (int dt = 0; dt < 4; ++dt) {
      const __bf16* vcol = Vt + headBase + (size_t)(dt * 16 + l15) * SEQ + kBase;
      v16bf bf = load_frag_k16(vcol, kg);
      cacc[dt] = wmma_bf16(ap, bf, cacc[dt]);
    }
  }

  // merge per-wave partials
  #pragma unroll
  for (int dt = 0; dt < 4; ++dt) {
    int d = dt * 16 + l15;
    #pragma unroll
    for (int j = 0; j < 8; ++j)
      atomicAdd(&ctx_acc[(mrow0 + j) * 64 + d], cacc[dt][j]);
  }
  __syncthreads();

  // ---- write fp32 output [B, S1, HIDDEN] ----
  #pragma unroll
  for (int i = 0; i < 4; ++i) {
    int idx = tid + i * 256;             // 16*64 outputs
    int m = idx >> 6, d = idx & 63;
    out[((size_t)b * SEQ + qT * 16 + m) * HIDDEN + h * HEAD_SIZE + d] = ctx_acc[idx];
  }
}

// =====================================================================
extern "C" void kernel_launch(void* const* d_in, const int* in_sizes, int n_in,
                              void* d_out, int out_size, void* d_ws, size_t ws_size,
                              hipStream_t stream) {
  const float* s1   = (const float*)d_in[0];
  const float* s2   = (const float*)d_in[1];
  const float* mask = (const float*)d_in[2];
  const float* Wq   = (const float*)d_in[3];
  const float* bq   = (const float*)d_in[4];
  const float* Wk   = (const float*)d_in[5];
  const float* bk   = (const float*)d_in[6];
  const float* Wv   = (const float*)d_in[7];
  const float* bv   = (const float*)d_in[8];
  const int*   cl   = (const int*)d_in[9];
  float* out = (float*)d_out;

  const size_t perTensor = (size_t)BATCH * NUM_HEADS * SEQ * HEAD_SIZE; // 8.39M
  const size_t actElems  = (size_t)BATCH * SEQ * HIDDEN;                // 8.39M
  const size_t wElems    = (size_t)HIDDEN * HIDDEN;                     // 1.05M
  __bf16* q_ws = (__bf16*)d_ws;
  __bf16* k_ws = q_ws + perTensor;
  __bf16* v_ws = k_ws + perTensor;
  __bf16* xb1  = v_ws + perTensor;
  __bf16* xb2  = xb1 + actElems;
  __bf16* wtq  = xb2 + actElems;
  __bf16* wtk  = wtq + wElems;
  __bf16* wtv  = wtk + wElems;

  // precision / layout pass
  dim3 bc(256);
  cvt_bf16_kernel<<<dim3((unsigned)(actElems / (256 * 8))), bc, 0, stream>>>(s1, xb1);
  cvt_bf16_kernel<<<dim3((unsigned)(actElems / (256 * 8))), bc, 0, stream>>>(s2, xb2);
  dim3 gt(HIDDEN / 32, HIDDEN / 32);
  transpose_cvt_kernel<<<gt, bc, 0, stream>>>(Wq, wtq);
  transpose_cvt_kernel<<<gt, bc, 0, stream>>>(Wk, wtk);
  transpose_cvt_kernel<<<gt, bc, 0, stream>>>(Wv, wtv);

  // projections (TDM-staged WMMA GEMMs)
  dim3 g1(HIDDEN / 128, (BATCH * SEQ) / 128);   // (8, 64)
  qkv_gemm_kernel<<<g1, bc, 0, stream>>>(xb1, wtq, bq, q_ws, HEAD_SIZE, 1);
  qkv_gemm_kernel<<<g1, bc, 0, stream>>>(xb2, wtk, bk, k_ws, HEAD_SIZE, 1);
  qkv_gemm_kernel<<<g1, bc, 0, stream>>>(xb2, wtv, bv, v_ws, 1, SEQ);

  // fused double-softmax attention
  dim3 g2(SEQ / 16, NUM_HEADS, BATCH);          // (64, 16, 8)
  co_attn_kernel<<<g2, bc, 0, stream>>>(q_ws, k_ws, v_ws, mask, cl, out);
}
